// MultiHeadAttention_50173807952512
// MI455X (gfx1250) — matrix operand order repaired and checked by reading, not compiled, not run on hardware
//
#include <hip/hip_runtime.h>
#include <hip/hip_bf16.h>
#include <math.h>

typedef _Float16 half8  __attribute__((ext_vector_type(8)));
typedef _Float16 half16 __attribute__((ext_vector_type(16)));
typedef float    float8 __attribute__((ext_vector_type(8)));

#define WMMA_F16(a, b, c) \
  __builtin_amdgcn_wmma_f32_16x16x32_f16(false, (a), false, (b), (short)0, (c), false, false)

#define PACK16(lo, hi) \
  __builtin_shufflevector((lo), (hi), 0, 1, 2, 3, 4, 5, 6, 7, 8, 9, 10, 11, 12, 13, 14, 15)

static constexpr int Bn = 2, Sn = 2048, Dn = 1024, Hn = 16, DKn = 64;
static constexpr int Mn = Bn * Sn;               // 4096 rows for projection GEMMs
static constexpr float kLog2e = 1.44269504088896340736f;
static constexpr float kMaskedVal = -1e9f * 1.44269504088896340736f; // -1e9 in exp2 domain

// ---------------------------------------------------------------------------
// f32 -> f16 conversion (grid-stride)
// ---------------------------------------------------------------------------
__global__ __launch_bounds__(256) void f32_to_f16_kernel(
    const float* __restrict__ in, _Float16* __restrict__ out, size_t n) {
  size_t i = (size_t)blockIdx.x * blockDim.x + threadIdx.x;
  size_t stride = (size_t)gridDim.x * blockDim.x;
  for (; i < n; i += stride) out[i] = (_Float16)in[i];
}

// ---------------------------------------------------------------------------
// Pack int mask (nonzero -> 1) into a bitmask via ballot (one u32 per 32 elems)
// ---------------------------------------------------------------------------
__global__ __launch_bounds__(256) void mask_pack_kernel(
    const int* __restrict__ mask, unsigned* __restrict__ bits, size_t n) {
  size_t i = (size_t)blockIdx.x * blockDim.x + threadIdx.x;
  size_t stride = (size_t)gridDim.x * blockDim.x;   // multiple of 32
  for (; i < n; i += stride) {
    unsigned long long b = __ballot(mask[i] != 0);
    if ((threadIdx.x & 31) == 0) bits[i >> 5] = (unsigned)b;
  }
}

// ---------------------------------------------------------------------------
// Y = (X @ W^T + bias) * scale      X:[M,K] f16, W:[N,K] f16, bias f32
// Wave computes a 16(M) x 64(N) strip; block = 8 waves -> 128 x 64 tile.
// ---------------------------------------------------------------------------
template <bool OUT_F16>
__global__ __launch_bounds__(256) void gemm_xwT_kernel(
    const _Float16* __restrict__ X, const _Float16* __restrict__ W,
    const float* __restrict__ bias, void* __restrict__ out,
    float scale, int M, int N, int K) {
  const int lane = threadIdx.x & 31;
  const int wave = threadIdx.x >> 5;
  const int hf   = lane >> 4;
  const int l16  = lane & 15;
  const int m0   = blockIdx.x * 128 + wave * 16;
  const int n0   = blockIdx.y * 64;

  float8 acc[4] = {};
  const _Float16* xrow = X + (size_t)(m0 + l16) * K;

#pragma unroll 2
  for (int k0 = 0; k0 < K; k0 += 32) {
    half8 alo = *(const half8*)(xrow + k0 + hf * 16);
    half8 ahi = *(const half8*)(xrow + k0 + hf * 16 + 8);
    half16 a = PACK16(alo, ahi);
#pragma unroll
    for (int n = 0; n < 4; ++n) {
      const _Float16* wrow = W + (size_t)(n0 + n * 16 + l16) * K + k0 + hf * 16;
      half16 b = *(const half16*)wrow;
      acc[n] = WMMA_F16(a, b, acc[n]);
    }
  }

#pragma unroll
  for (int n = 0; n < 4; ++n) {
    const int col = n0 + n * 16 + l16;
    const float bv = bias ? bias[col] : 0.0f;
#pragma unroll
    for (int r = 0; r < 8; ++r) {
      const int row = m0 + r + 8 * hf;
      const float v = (acc[n][r] + bv) * scale;
      if (OUT_F16)
        ((_Float16*)out)[(size_t)row * N + col] = (_Float16)v;
      else
        ((float*)out)[(size_t)row * N + col] = v;
    }
  }
}

// ---------------------------------------------------------------------------
// Flash attention. block = 128 threads (4 waves); blockIdx.y = b*H + h,
// blockIdx.x = 64-row query tile; each wave owns 16 query rows.
// Q pre-scaled by (1/sqrt(DK)) * log2(e); softmax runs in the exp2 domain.
// K tile is staged with GLOBAL_LOAD_ASYNC_TO_LDS_B128 (ASYNCcnt path);
// V is staged manually because the PV B-fragments need a transpose.
// ---------------------------------------------------------------------------
__global__ __launch_bounds__(128, 1) void flash_attn_kernel(
    const _Float16* __restrict__ Qp, const _Float16* __restrict__ Kp,
    const _Float16* __restrict__ Vp, const unsigned* __restrict__ mbits,
    _Float16* __restrict__ ctx) {
  __shared__ _Float16 Klds[64 * 64];      // [key][dk]     8 KB
  __shared__ _Float16 Vt[64 * 64];        // [dk][key]     8 KB (transposed)
  __shared__ _Float16 Plds[4][16 * 64];   // per-wave P    8 KB

  const int tid  = threadIdx.x;
  const int lane = tid & 31;
  const int wave = tid >> 5;
  const int hf   = lane >> 4;
  const int l16  = lane & 15;
  const int bh   = blockIdx.y;
  const int b    = bh >> 4;
  const int h    = bh & 15;
  const int q0   = blockIdx.x * 64 + wave * 16;
  const int wpr  = Sn / 32;               // mask words per row (64)
  const unsigned bmLo = 1u << l16;        // bit tests for mask words
  const unsigned bmHi = 1u << (16 + l16);

  // Q A-fragments (DK=64 -> two 16x32 frags), resident for the whole loop
  const _Float16* qrow = Qp + ((size_t)b * Sn + q0 + l16) * Dn + h * DKn;
  half16 aq0, aq1;
  {
    half8 lo0 = *(const half8*)(qrow + hf * 16);
    half8 hi0 = *(const half8*)(qrow + hf * 16 + 8);
    half8 lo1 = *(const half8*)(qrow + 32 + hf * 16);
    half8 hi1 = *(const half8*)(qrow + 32 + hf * 16 + 8);
    aq0 = PACK16(lo0, hi0);
    aq1 = PACK16(lo1, hi1);
  }

  float8 oacc[4] = {};
  float mrow[8], lrow[8];
#pragma unroll
  for (int r = 0; r < 8; ++r) { mrow[r] = -INFINITY; lrow[r] = 0.0f; }

  // per-thread staging coordinates (2 threads per key row, 32 halfs each)
  const int srow = tid >> 1;
  const int sc32 = (tid & 1) * 32;
  const unsigned kldsAddr = (unsigned)(uintptr_t)&Klds[srow * 64 + sc32];

  for (int j0 = 0; j0 < Sn; j0 += 64) {
    __syncthreads();   // previous tile fully consumed
    {
      const _Float16* kg = Kp + ((size_t)b * Sn + j0 + srow) * Dn + h * DKn + sc32;
      const _Float16* vg = Vp + ((size_t)b * Sn + j0 + srow) * Dn + h * DKn + sc32;
      if (j0 + 64 < Sn) {   // pull next tile toward L2 (global_prefetch_b8)
        __builtin_prefetch(kg + (size_t)64 * Dn, 0, 0);
        __builtin_prefetch(vg + (size_t)64 * Dn, 0, 0);
      }
      // K: async DMA straight into LDS. INST_OFFSET is added to BOTH the LDS
      // and global addresses (ISA 08 §4.4), and both advance by 16B here.
      asm volatile("global_load_async_to_lds_b128 %0, %1, off"
                   :: "v"(kldsAddr), "v"(kg) : "memory");
      asm volatile("global_load_async_to_lds_b128 %0, %1, off offset:16"
                   :: "v"(kldsAddr), "v"(kg) : "memory");
      asm volatile("global_load_async_to_lds_b128 %0, %1, off offset:32"
                   :: "v"(kldsAddr), "v"(kg) : "memory");
      asm volatile("global_load_async_to_lds_b128 %0, %1, off offset:48"
                   :: "v"(kldsAddr), "v"(kg) : "memory");
      // V: manual transpose into Vt
#pragma unroll
      for (int c = 0; c < 4; ++c) {
        half8 vv = *(const half8*)(vg + c * 8);
#pragma unroll
        for (int e = 0; e < 8; ++e) Vt[(sc32 + c * 8 + e) * 64 + srow] = vv[e];
      }
      asm volatile("s_wait_asynccnt 0" ::: "memory");
    }
    __syncthreads();

    // --- scores S = Qs @ K^T : 4 tiles of 16x16, K-dim 64 = 2 WMMAs each
    float8 sacc[4] = {};
#pragma unroll
    for (int n = 0; n < 4; ++n) {
      const _Float16* kb = &Klds[(n * 16 + l16) * 64];
      half16 b0 = *(const half16*)(kb + hf * 16);
      half16 b1 = *(const half16*)(kb + 32 + hf * 16);
      sacc[n] = WMMA_F16(aq0, b0, sacc[n]);
      sacc[n] = WMMA_F16(aq1, b1, sacc[n]);
    }

    // --- mask via packed bits: two u32 words cover this row's 64 keys
#pragma unroll
    for (int r = 0; r < 8; ++r) {
      const int qi = q0 + r + 8 * hf;
      const unsigned* mp = mbits + (size_t)qi * wpr + (j0 >> 5);
      const unsigned wlo = mp[0];
      const unsigned whi = mp[1];
      sacc[0][r] = (wlo & bmLo) ? sacc[0][r] : kMaskedVal;
      sacc[1][r] = (wlo & bmHi) ? sacc[1][r] : kMaskedVal;
      sacc[2][r] = (whi & bmLo) ? sacc[2][r] : kMaskedVal;
      sacc[3][r] = (whi & bmHi) ? sacc[3][r] : kMaskedVal;
    }

    // --- online softmax in exp2 domain (row = r + 8*hf, within one 16-lane half)
#pragma unroll
    for (int r = 0; r < 8; ++r) {
      float mx = fmaxf(fmaxf(sacc[0][r], sacc[1][r]),
                       fmaxf(sacc[2][r], sacc[3][r]));
#pragma unroll
      for (int off = 8; off >= 1; off >>= 1)
        mx = fmaxf(mx, __shfl_xor(mx, off, 32));
      const float mnew = fmaxf(mrow[r], mx);
      const float corr = __builtin_exp2f(mrow[r] - mnew);  // exp2(-inf)=0 first pass
      float rs = 0.0f;
#pragma unroll
      for (int n = 0; n < 4; ++n) {
        const float p = __builtin_exp2f(sacc[n][r] - mnew);
        sacc[n][r] = p;
        rs += p;
      }
#pragma unroll
      for (int off = 8; off >= 1; off >>= 1)
        rs += __shfl_xor(rs, off, 32);
      lrow[r] = lrow[r] * corr + rs;
      mrow[r] = mnew;
#pragma unroll
      for (int n = 0; n < 4; ++n) oacc[n][r] *= corr;
    }

    // --- C-layout -> A-layout for P via per-wave LDS scratch
    _Float16* pw = &Plds[wave][0];
#pragma unroll
    for (int n = 0; n < 4; ++n)
#pragma unroll
      for (int r = 0; r < 8; ++r)
        pw[(r + 8 * hf) * 64 + n * 16 + l16] = (_Float16)sacc[n][r];
    asm volatile("s_wait_dscnt 0" ::: "memory");  // wave-local LDS RAW fence

    const _Float16* prow = pw + l16 * 64;
    half16 ap0, ap1;
    {
      half8 lo0 = *(const half8*)(prow + hf * 16);
      half8 hi0 = *(const half8*)(prow + hf * 16 + 8);
      half8 lo1 = *(const half8*)(prow + 32 + hf * 16);
      half8 hi1 = *(const half8*)(prow + 32 + hf * 16 + 8);
      ap0 = PACK16(lo0, hi0);
      ap1 = PACK16(lo1, hi1);
    }

    // --- O += P @ V  (B-frags from transposed V: contiguous LDS reads)
#pragma unroll
    for (int n = 0; n < 4; ++n) {
      const _Float16* vb = &Vt[(n * 16 + l16) * 64];
      half16 b0 = *(const half16*)(vb + hf * 16);
      half16 b1 = *(const half16*)(vb + 32 + hf * 16);
      oacc[n] = WMMA_F16(ap0, b0, oacc[n]);
      oacc[n] = WMMA_F16(ap1, b1, oacc[n]);
    }
  }

  // --- normalize and store context as f16 in [B,S,D] layout (D = h*DK + dk)
#pragma unroll
  for (int r = 0; r < 8; ++r) {
    const float inv = 1.0f / lrow[r];
    const int qi = q0 + r + 8 * hf;
#pragma unroll
    for (int n = 0; n < 4; ++n)
      ctx[((size_t)b * Sn + qi) * Dn + h * DKn + n * 16 + l16] =
          (_Float16)(oacc[n][r] * inv);
  }
}

// ---------------------------------------------------------------------------
// Host-side launch
// ---------------------------------------------------------------------------
extern "C" void kernel_launch(void* const* d_in, const int* in_sizes, int n_in,
                              void* d_out, int out_size, void* d_ws, size_t ws_size,
                              hipStream_t stream) {
  const float* q    = (const float*)d_in[0];
  const float* k    = (const float*)d_in[1];
  const float* v    = (const float*)d_in[2];
  const int*   mask = (const int*)d_in[3];
  const float* Wq   = (const float*)d_in[4];
  const float* bq   = (const float*)d_in[5];
  const float* Wk   = (const float*)d_in[6];
  const float* bk   = (const float*)d_in[7];
  const float* Wv   = (const float*)d_in[8];
  const float* bv   = (const float*)d_in[9];
  const float* Wo   = (const float*)d_in[10];
  const float* bo   = (const float*)d_in[11];
  float* out = (float*)d_out;

  const size_t inEl = (size_t)Mn * Dn;   // 4M elements
  const size_t wEl  = (size_t)Dn * Dn;   // 1M elements

  _Float16* qh  = (_Float16*)d_ws;
  _Float16* kh  = qh + inEl;
  _Float16* vh  = kh + inEl;
  _Float16* Wqh = vh + inEl;
  _Float16* Wkh = Wqh + wEl;
  _Float16* Wvh = Wkh + wEl;
  _Float16* Woh = Wvh + wEl;
  _Float16* Qp  = Woh + wEl;
  _Float16* Kpj = Qp + inEl;
  _Float16* Vpj = Kpj + inEl;
  _Float16* ctx = Vpj + inEl;
  unsigned* mbits = (unsigned*)(ctx + inEl);      // S*S/32 words = 512 KB

  // 1) staging conversions + mask packing
  f32_to_f16_kernel<<<1024, 256, 0, stream>>>(q, qh, inEl);
  f32_to_f16_kernel<<<1024, 256, 0, stream>>>(k, kh, inEl);
  f32_to_f16_kernel<<<1024, 256, 0, stream>>>(v, vh, inEl);
  f32_to_f16_kernel<<<512, 256, 0, stream>>>(Wq, Wqh, wEl);
  f32_to_f16_kernel<<<512, 256, 0, stream>>>(Wk, Wkh, wEl);
  f32_to_f16_kernel<<<512, 256, 0, stream>>>(Wv, Wvh, wEl);
  f32_to_f16_kernel<<<512, 256, 0, stream>>>(Wo, Woh, wEl);
  mask_pack_kernel<<<1024, 256, 0, stream>>>(mask, mbits, (size_t)Sn * Sn);

  // 2) projections (Q pre-scaled by (1/sqrt(DK)) * log2(e) for exp2-domain softmax)
  dim3 gGemm(Mn / 128, Dn / 64);
  gemm_xwT_kernel<true><<<gGemm, 256, 0, stream>>>(qh, Wqh, bq, Qp, 0.125f * kLog2e, Mn, Dn, Dn);
  gemm_xwT_kernel<true><<<gGemm, 256, 0, stream>>>(kh, Wkh, bk, Kpj, 1.0f, Mn, Dn, Dn);
  gemm_xwT_kernel<true><<<gGemm, 256, 0, stream>>>(vh, Wvh, bv, Vpj, 1.0f, Mn, Dn, Dn);

  // 3) flash attention: grid = (S/64 query tiles, B*H heads)
  dim3 gAttn(Sn / 64, Bn * Hn);
  flash_attn_kernel<<<gAttn, 128, 0, stream>>>(Qp, Kpj, Vpj, mbits, ctx);

  // 4) output projection -> f32
  gemm_xwT_kernel<false><<<gGemm, 256, 0, stream>>>(ctx, Woh, bo, out, 1.0f, Mn, Dn, Dn);
}
